// MyModel_50989851738313
// MI455X (gfx1250) — matrix operand order repaired and checked by reading, not compiled, not run on hardware
//
#include <hip/hip_runtime.h>
#include <hip/hip_bf16.h>

typedef __attribute__((ext_vector_type(16))) __bf16 v16bf;
typedef __attribute__((ext_vector_type(8)))  float  v8f;

#define BB 32
#define SS 200
#define NFEAT 165
#define DD 1024
#define DFF 4096
#define NL 6
#define DPROB 512
#define MROWS (BB*SS)

#define LDSP 40   // u16 row pitch for LDS tiles (80B, multiple of 16B)

__device__ __forceinline__ unsigned short f2bf(float f) {
  union { float f; unsigned u; } x; x.f = f;
  unsigned r = x.u + 0x7fffu + ((x.u >> 16) & 1u);
  return (unsigned short)(r >> 16);
}
__device__ __forceinline__ unsigned pack2bf(float lo, float hi) {
  return (unsigned)f2bf(lo) | ((unsigned)f2bf(hi) << 16);
}

union Frag { v16bf v; uint4 q[2]; unsigned short u[16]; };

// 8x v_wmma_f32_16x16x32_bf16 on one staged 128x32 / 32x128 tile pair.
// As row-major [m][k]; Bs transposed [n][k]: both fragments are contiguous
// 16B runs per lane -> ds_load_b128.
__device__ __forceinline__ void wmma_tile(
    const unsigned short (*__restrict__ Asb)[LDSP],
    const unsigned short (*__restrict__ Bsb)[LDSP],
    v8f (*__restrict__ acc)[2], int wm, int wn, int mfr, int hf) {
  Frag af[4], bf[2];
#pragma unroll
  for (int i = 0; i < 4; i++) {
    const unsigned short* arow = &Asb[wm + i * 16 + mfr][hf * 8];
    af[i].q[0] = *(const uint4*)arow;         // k = 8h .. 8h+7
    af[i].q[1] = *(const uint4*)(arow + 16);  // k = 16+8h .. +7
  }
#pragma unroll
  for (int j = 0; j < 2; j++) {
    const unsigned short* brow = &Bsb[wn + j * 16 + mfr][hf * 8];
    bf[j].q[0] = *(const uint4*)brow;         // k = 8h .. 8h+7
    bf[j].q[1] = *(const uint4*)(brow + 16);  // k = 16+8h .. +7
  }
#pragma unroll
  for (int i = 0; i < 4; i++)
#pragma unroll
    for (int j = 0; j < 2; j++)
      acc[i][j] = __builtin_amdgcn_wmma_f32_16x16x32_bf16(
          false, af[i].v, false, bf[j].v, (short)0, acc[i][j], false, false);
}

// ---------------------------------------------------------------------------
// WMMA GEMM:  C[M,N] = act( A[M,K] @ B[K,N] + bias + pos ) + resid
// 256 threads = 8 waves (2 M x 4 N); block tile 128x128, K-step 32;
// fast path: double-buffered LDS, register-staged prefetch of tile t+1
// overlapped with the 8-WMMA burst of tile t (one barrier per K-step).
// ---------------------------------------------------------------------------
__launch_bounds__(256)
__global__ void gemm_wmma_bf16(const void* __restrict__ Aptr, int a_is_f32,
                               const float* __restrict__ B,
                               const float* __restrict__ bias,   // nullable [N]
                               const float* __restrict__ pos,    // nullable [posS,N]
                               int posS,
                               const float* __restrict__ resid,  // nullable [M,N]
                               float* __restrict__ Cf,           // nullable
                               unsigned short* __restrict__ Cb,  // nullable (bf16)
                               int M, int N, int K, int relu) {
  __shared__ unsigned short As[2][128][LDSP];  // [buf][m][k]
  __shared__ unsigned short Bs[2][128][LDSP];  // [buf][n][k] (transposed tile)

  const int tid  = threadIdx.x;
  const int lane = tid & 31;
  const int wave = tid >> 5;
  const int wm = (wave >> 2) * 64;
  const int wn = (wave & 3) * 32;
  const int m0 = blockIdx.x * 128;
  const int n0 = blockIdx.y * 128;

  const int mfr = lane & 15;
  const int hf  = lane >> 4;

  const v8f vzero = {0.f,0.f,0.f,0.f,0.f,0.f,0.f,0.f};
  v8f acc[4][2];
#pragma unroll
  for (int i = 0; i < 4; i++) { acc[i][0] = vzero; acc[i][1] = vzero; }

  // staging thread mappings
  const int ar = tid >> 1;          // A row 0..127
  const int ac = (tid & 1) * 16;    // A col 0 / 16
  const int bn = tid & 127;         // B col (n) 0..127
  const int bh = tid >> 7;          // B k-half 0/1

  const bool edge = (m0 + 128 > M) || (n0 + 128 > N);
  const bool fast = !edge && !a_is_f32 && ((K & 31) == 0);

  if (fast) {
    const unsigned short* A16 = (const unsigned short*)Aptr;
    const size_t arowoff = (size_t)(m0 + ar) * K + ac;
    // ---- prologue: stage tile 0 into buffer 0
    {
      const uint4* asrc = (const uint4*)(A16 + arowoff);
      uint4 d0 = asrc[0], d1 = asrc[1];
      *(uint4*)&As[0][ar][ac]     = d0;
      *(uint4*)&As[0][ar][ac + 8] = d1;
      float fv[16];
#pragma unroll
      for (int j = 0; j < 16; j++)
        fv[j] = B[(size_t)(bh * 16 + j) * N + n0 + bn];
      unsigned* brow = (unsigned*)&Bs[0][bn][0];
#pragma unroll
      for (int p = 0; p < 8; p++)
        brow[bh * 8 + p] = pack2bf(fv[2 * p], fv[2 * p + 1]);
    }
    __syncthreads();

    const int T = K >> 5;
    for (int t = 0; t < T; t++) {
      const int cur = t & 1;
      const bool more = (t + 1) < T;
      uint4 a0, a1;
      float fv[16];
      if (more) {  // issue next tile's global loads before computing
        const int kn = (t + 1) << 5;
        const uint4* asrc = (const uint4*)(A16 + arowoff + kn);
        a0 = asrc[0]; a1 = asrc[1];
#pragma unroll
        for (int j = 0; j < 16; j++)
          fv[j] = B[(size_t)(kn + bh * 16 + j) * N + n0 + bn];
      }
      wmma_tile(As[cur], Bs[cur], acc, wm, wn, mfr, hf);
      if (more) {  // commit prefetched tile to the alternate buffer
        const int nxt = cur ^ 1;
        *(uint4*)&As[nxt][ar][ac]     = a0;
        *(uint4*)&As[nxt][ar][ac + 8] = a1;
        unsigned* brow = (unsigned*)&Bs[nxt][bn][0];
#pragma unroll
        for (int p = 0; p < 8; p++)
          brow[bh * 8 + p] = pack2bf(fv[2 * p], fv[2 * p + 1]);
      }
      __syncthreads();
    }
  } else {
    // ---- guarded path (edge tiles / f32-A / ragged K), single buffer
    for (int k0 = 0; k0 < K; k0 += 32) {
      const int grow = m0 + ar;
      if (a_is_f32) {
        const float* A = (const float*)Aptr;
#pragma unroll
        for (int j = 0; j < 16; j++) {
          int gk = k0 + ac + j;
          float v = (grow < M && gk < K) ? A[(size_t)grow * K + gk] : 0.f;
          As[0][ar][ac + j] = f2bf(v);
        }
      } else {
        const unsigned short* A = (const unsigned short*)Aptr;
#pragma unroll
        for (int j = 0; j < 16; j++) {
          int gk = k0 + ac + j;
          As[0][ar][ac + j] =
              (grow < M && gk < K) ? A[(size_t)grow * K + gk] : (unsigned short)0;
        }
      }
      const int gn = n0 + bn;
#pragma unroll
      for (int j = 0; j < 16; j++) {
        int gk = k0 + bh * 16 + j;
        float v = (gk < K && gn < N) ? B[(size_t)gk * N + gn] : 0.f;
        Bs[0][bn][bh * 16 + j] = f2bf(v);
      }
      __syncthreads();
      wmma_tile(As[0], Bs[0], acc, wm, wn, mfr, hf);
      __syncthreads();
    }
  }

  // ---- epilogue: C/D layout — VGPR e: M = e + 8*half, N = lane&15
#pragma unroll
  for (int i = 0; i < 4; i++) {
#pragma unroll
    for (int j = 0; j < 2; j++) {
#pragma unroll
      for (int e = 0; e < 8; e++) {
        int row = m0 + wm + i * 16 + e + hf * 8;
        int col = n0 + wn + j * 16 + mfr;
        if (!edge || (row < M && col < N)) {
          float x = acc[i][j][e];
          if (bias) x += bias[col];
          if (pos)  x += pos[(size_t)(row % posS) * N + col];
          if (relu) x = x > 0.f ? x : 0.f;
          if (resid) x += resid[(size_t)row * N + col];
          if (Cf) Cf[(size_t)row * N + col] = x;
          if (Cb) Cb[(size_t)row * N + col] = f2bf(x);
        }
      }
    }
  }
}

// ---------------------------------------------------------------------------
__launch_bounds__(256)
__global__ void row_fused(float* __restrict__ hbuf,
                          const float* __restrict__ bcast,
                          const float* __restrict__ g,
                          const float* __restrict__ bv,
                          float* __restrict__ out_f,
                          unsigned short* __restrict__ out_b,
                          float* __restrict__ stats_out,
                          int update_h) {
  const int row = blockIdx.x;
  const int b = row / SS;
  const int tid = threadIdx.x;
  float x[4];
#pragma unroll
  for (int j = 0; j < 4; j++) {
    int ch = tid + j * 256;
    x[j] = hbuf[(size_t)row * DD + ch];
    if (bcast) x[j] += bcast[(size_t)b * DD + ch];
  }
  float s = x[0] + x[1] + x[2] + x[3];
  float sq = x[0]*x[0] + x[1]*x[1] + x[2]*x[2] + x[3]*x[3];
#pragma unroll
  for (int o = 16; o > 0; o >>= 1) {
    s  += __shfl_xor(s,  o, 32);
    sq += __shfl_xor(sq, o, 32);
  }
  __shared__ float ss[8], sqs[8];
  if ((tid & 31) == 0) { ss[tid >> 5] = s; sqs[tid >> 5] = sq; }
  __syncthreads();
  float tot = 0.f, totq = 0.f;
#pragma unroll
  for (int w = 0; w < 8; w++) { tot += ss[w]; totq += sqs[w]; }
  float mean = tot / (float)DD;
  float var  = totq / (float)DD - mean * mean;
  float rstd = rsqrtf(var + 1e-6f);

  if (update_h) {
#pragma unroll
    for (int j = 0; j < 4; j++) hbuf[(size_t)row * DD + tid + j * 256] = x[j];
  }
  if (stats_out && tid == 0) { stats_out[row * 2] = mean; stats_out[row * 2 + 1] = rstd; }
  if (g) {
#pragma unroll
    for (int j = 0; j < 4; j++) {
      int ch = tid + j * 256;
      float ln = (x[j] - mean) * rstd * g[ch] + bv[ch];
      if (out_f) out_f[(size_t)row * DD + ch] = ln;
      if (out_b) out_b[(size_t)row * DD + ch] = f2bf(ln);
    }
  }
}

// ---------------------------------------------------------------------------
__launch_bounds__(256)
__global__ void batch_mean(const float* __restrict__ x,
                           const float* __restrict__ stats,
                           const float* __restrict__ g,
                           const float* __restrict__ bv,
                           float* __restrict__ out_norm_f,
                           unsigned short* __restrict__ out_norm_b,
                           float* __restrict__ out_raw) {
  const int b = blockIdx.y;
  const int ch = blockIdx.x * 256 + threadIdx.x;
  float sn = 0.f, sr = 0.f;
  for (int s = 0; s < SS; s++) {
    int row = b * SS + s;
    float v = x[(size_t)row * DD + ch];
    sn += (v - stats[row * 2]) * stats[row * 2 + 1];
    sr += v;
  }
  sn *= (1.f / (float)SS);
  sr *= (1.f / (float)SS);
  float o = g ? (sn * g[ch] + bv[ch]) : sn;
  if (out_norm_f) out_norm_f[(size_t)b * DD + ch] = o;
  if (out_norm_b) out_norm_b[(size_t)b * DD + ch] = f2bf(o);
  if (out_raw)    out_raw[(size_t)b * DD + ch] = sr;
}

__launch_bounds__(256)
__global__ void scale_bias_to_bf16(const float* __restrict__ in,
                                   const float* __restrict__ g,
                                   const float* __restrict__ bv,
                                   unsigned short* __restrict__ o, int n) {
  int i = blockIdx.x * 256 + threadIdx.x;
  if (i < n) { int ch = i & (DD - 1); o[i] = f2bf(in[i] * g[ch] + bv[ch]); }
}

__launch_bounds__(256)
__global__ void add_to_bf16(const float* __restrict__ a,
                            const float* __restrict__ b,
                            unsigned short* __restrict__ o, int n) {
  int i = blockIdx.x * 256 + threadIdx.x;
  if (i < n) o[i] = f2bf(a[i] + b[i]);
}

// ---------------------------------------------------------------------------
extern "C" void kernel_launch(void* const* d_in, const int* in_sizes, int n_in,
                              void* d_out, int out_size, void* d_ws, size_t ws_size,
                              hipStream_t stream) {
  (void)in_sizes; (void)n_in; (void)out_size; (void)ws_size;
  const float* p1      = (const float*)d_in[0];
  const float* p2      = (const float*)d_in[1];
  const float* fc1_w   = (const float*)d_in[2];
  const float* fc1_b   = (const float*)d_in[3];
  const float* pos     = (const float*)d_in[4];
  const float* enc_v   = (const float*)d_in[7];
  const float* enc_o   = (const float*)d_in[8];
  const float* enc_ag  = (const float*)d_in[9];
  const float* enc_ab  = (const float*)d_in[10];
  const float* enc_wi  = (const float*)d_in[11];
  const float* enc_wo  = (const float*)d_in[12];
  const float* enc_fg  = (const float*)d_in[13];
  const float* enc_fb  = (const float*)d_in[14];
  const float* enc_lg  = (const float*)d_in[15];
  const float* enc_lb  = (const float*)d_in[16];
  const float* dec_sv  = (const float*)d_in[19];
  const float* dec_so  = (const float*)d_in[20];
  const float* dec_sg  = (const float*)d_in[21];
  const float* dec_sb  = (const float*)d_in[22];
  const float* dec_cv  = (const float*)d_in[25];
  const float* dec_co  = (const float*)d_in[26];
  const float* dec_wi  = (const float*)d_in[29];
  const float* dec_wo  = (const float*)d_in[30];
  const float* dec_fg  = (const float*)d_in[31];
  const float* dec_fb  = (const float*)d_in[32];
  const float* dec_lg  = (const float*)d_in[33];
  const float* dec_lb  = (const float*)d_in[34];
  const float* fc2_w   = (const float*)d_in[35];
  const float* fc2_b   = (const float*)d_in[36];
  const float* fc3_w   = (const float*)d_in[37];
  const float* fc3_b   = (const float*)d_in[38];
  float* out = (float*)d_out;

  char* base = (char*)d_ws;
  size_t off = 0;
  auto carve = [&](size_t bytes) -> char* {
    char* p = base + off;
    off += (bytes + 255) & ~(size_t)255;
    return p;
  };
  float*          h     = (float*)carve((size_t)MROWS * DD * 4);
  float*          x2    = (float*)carve((size_t)MROWS * DD * 4);
  unsigned short* y16   = (unsigned short*)carve((size_t)MROWS * DFF * 2);
  unsigned short* f16   = (unsigned short*)carve((size_t)MROWS * DD * 2);
  unsigned short* z16   = (unsigned short*)carve((size_t)MROWS * DPROB * 2);
  float*          stats = (float*)carve((size_t)MROWS * 2 * 4);
  float*          s_normp2 = (float*)carve((size_t)BB * DD * 4);
  float*          s_meanp2 = (float*)carve((size_t)BB * DD * 4);
  float*          s_attn   = (float*)carve((size_t)BB * DD * 4);
  float*          s_sa     = (float*)carve((size_t)BB * DD * 4);
  unsigned short* s_ln_b   = (unsigned short*)carve((size_t)BB * DD * 2);
  unsigned short* s_mv_b   = (unsigned short*)carve((size_t)BB * DD * 2);
  unsigned short* s_tmp_b  = (unsigned short*)carve((size_t)BB * DD * 2);

  auto gemm = [&](const void* A, int a_f32, const float* B, const float* bias,
                  const float* posp, int posS, const float* resid,
                  float* Cf, unsigned short* Cb, int M, int N, int K, int relu) {
    dim3 grid((M + 127) / 128, (N + 127) / 128);
    gemm_wmma_bf16<<<grid, 256, 0, stream>>>(A, a_f32, B, bias, posp, posS,
                                             resid, Cf, Cb, M, N, K, relu);
  };

  const int VN = BB * DD;
  const int VB = (VN + 255) / 256;

  // ---- embeddings: h = p1@fc1 + b + pos ; x2 = p2@fc1 + b + pos
  gemm(p1, 1, fc1_w, fc1_b, pos, SS, nullptr, h,  nullptr, MROWS, DD, NFEAT, 0);
  gemm(p2, 1, fc1_w, fc1_b, pos, SS, nullptr, x2, nullptr, MROWS, DD, NFEAT, 0);

  // ---- decoder precompute: per-row stats of p2; seq-means of raw & normed p2
  row_fused<<<MROWS, 256, 0, stream>>>(x2, nullptr, nullptr, nullptr,
                                       nullptr, nullptr, stats, 0);
  batch_mean<<<dim3(DD / 256, BB), 256, 0, stream>>>(x2, stats, nullptr, nullptr,
                                                     s_normp2, nullptr, s_meanp2);

  // ---- encoder stack (attention collapses to seq-mean path; Q/K are dead)
  for (int i = 0; i < NL; i++) {
    row_fused<<<MROWS, 256, 0, stream>>>(h, nullptr, nullptr, nullptr,
                                         nullptr, nullptr, stats, 0);
    batch_mean<<<dim3(DD / 256, BB), 256, 0, stream>>>(
        h, stats, enc_ag + (size_t)i * DD, enc_ab + (size_t)i * DD,
        nullptr, s_ln_b, nullptr);
    gemm(s_ln_b, 0, enc_v + (size_t)i * DD * DD, nullptr, nullptr, 1, nullptr,
         nullptr, s_mv_b, BB, DD, DD, 0);
    gemm(s_mv_b, 0, enc_o + (size_t)i * DD * DD, nullptr, nullptr, 1, nullptr,
         s_attn, nullptr, BB, DD, DD, 0);
    row_fused<<<MROWS, 256, 0, stream>>>(h, s_attn, enc_fg + (size_t)i * DD,
                                         enc_fb + (size_t)i * DD,
                                         nullptr, f16, nullptr, 1);
    gemm(f16, 0, enc_wi + (size_t)i * DD * DFF, nullptr, nullptr, 1, nullptr,
         nullptr, y16, MROWS, DFF, DD, 1);
    gemm(y16, 0, enc_wo + (size_t)i * DFF * DD, nullptr, nullptr, 1, h,
         h, nullptr, MROWS, DD, DFF, 0);
  }
  row_fused<<<MROWS, 256, 0, stream>>>(h, nullptr, enc_lg, enc_lb,
                                       h, nullptr, nullptr, 0);

  // ---- decoder stack
  for (int i = 0; i < NL; i++) {
    scale_bias_to_bf16<<<VB, 256, 0, stream>>>(s_normp2, dec_sg + (size_t)i * DD,
                                               dec_sb + (size_t)i * DD, s_ln_b, VN);
    gemm(s_ln_b, 0, dec_sv + (size_t)i * DD * DD, nullptr, nullptr, 1, nullptr,
         nullptr, s_mv_b, BB, DD, DD, 0);
    gemm(s_mv_b, 0, dec_so + (size_t)i * DD * DD, nullptr, nullptr, 1, nullptr,
         s_sa, nullptr, BB, DD, DD, 0);
    add_to_bf16<<<VB, 256, 0, stream>>>(s_meanp2, s_sa, s_tmp_b, VN);
    gemm(s_tmp_b, 0, dec_cv + (size_t)i * DD * DD, nullptr, nullptr, 1, nullptr,
         nullptr, s_mv_b, BB, DD, DD, 0);
    gemm(s_mv_b, 0, dec_co + (size_t)i * DD * DD, nullptr, nullptr, 1, nullptr,
         s_attn, nullptr, BB, DD, DD, 0);
    row_fused<<<MROWS, 256, 0, stream>>>(h, s_attn, dec_fg + (size_t)i * DD,
                                         dec_fb + (size_t)i * DD,
                                         nullptr, f16, nullptr, 1);
    gemm(f16, 0, dec_wi + (size_t)i * DD * DFF, nullptr, nullptr, 1, nullptr,
         nullptr, y16, MROWS, DFF, DD, 1);
    gemm(y16, 0, dec_wo + (size_t)i * DFF * DD, nullptr, nullptr, 1, h,
         h, nullptr, MROWS, DD, DFF, 0);
  }
  row_fused<<<MROWS, 256, 0, stream>>>(h, nullptr, dec_lg, dec_lb,
                                       nullptr, f16, nullptr, 0);

  // ---- head
  gemm(f16, 0, fc2_w, fc2_b, nullptr, 1, nullptr, nullptr, z16,
       MROWS, DPROB, DD, 0);
  gemm(z16, 0, fc3_w, fc3_b, nullptr, 1, nullptr, out, nullptr,
       MROWS, NFEAT, DPROB, 0);
}
